// Attention_41472204210330
// MI455X (gfx1250) — compile-verified, matrix operand order checked by reading, not run on hardware
//
#include <hip/hip_runtime.h>

// ---------------------------------------------------------------------------
// Attention block for MI455X (gfx1250, wave32, WMMA).
// - All GEMMs bf16 via v_wmma_f32_16x16x32_bf16, f32 accumulation.
// - Flash-attention streaming softmax; scores never hit HBM.
// - softmax scale (1/sqrt(64)) folded into RoPE'd Q -> no scale VALU in the
//   attention inner loop.
// - V projection written transposed ([B][H][D][T]) so attention reads V
//   B-fragments straight from global memory (contiguous along keys).
// - K tiles double-buffered in LDS via global_load_async_to_lds_b128
//   (ASYNCcnt), overlapping fetch of tile i+1 with WMMAs of tile i.
// - GEMM weight panels staged to LDS with async-to-LDS as well.
// ---------------------------------------------------------------------------

#define HID   1024
#define NHEAD 16
#define HDIM  64
#define BATCH 2
#define SEQ   2048
#define NTOK  (BATCH * SEQ)

typedef unsigned short u16;
typedef __attribute__((ext_vector_type(16))) __bf16 v16bf;
typedef __attribute__((ext_vector_type(8)))  float  v8f;

union Frag {           // 8 VGPRs = 16 bf16 (one WMMA A or B operand, wave32)
    uint4 u4[2];
    v16bf v;
};

__device__ __forceinline__ u16 f2bf(float f) {      // round-to-nearest-even
    unsigned u = __float_as_uint(f);
    unsigned r = u + 0x7FFFu + ((u >> 16) & 1u);
    return (u16)(r >> 16);
}
__device__ __forceinline__ float bf2f(u16 h) {
    return __uint_as_float(((unsigned)h) << 16);
}

// ---------------------------------------------------------------------------
// 1) f32 -> bf16 conversion
// ---------------------------------------------------------------------------
__global__ __launch_bounds__(256) void cvt_f32_bf16(const float* __restrict__ src,
                                                    u16* __restrict__ dst, int n) {
    int i = blockIdx.x * blockDim.x + threadIdx.x;
    int stride = gridDim.x * blockDim.x;
    for (; i < n; i += stride) dst[i] = f2bf(src[i]);
}

// ---------------------------------------------------------------------------
// 2) C[M,1024] = A[M,1024](bf16) x W[1024,1024]^T(bf16), f32 accum.
//    Block: 8 waves -> 128 rows x 16 cols.  B-panel staged to LDS via
//    async-to-LDS with +8 u16 pad per row -> conflict-free ds_load_b128.
//    Epilogue variants: f32 row-major, bf16 row-major, bf16 transposed-per-
//    head ([B][H][D][T], used for V so attention needs no V transpose).
// ---------------------------------------------------------------------------
#define BSTRIDE 1032   // 1024 + 8 u16 pad

#define OUTK_F32     0
#define OUTK_BF16    1
#define OUTK_BF16_VT 2

template <int OUTK>
__global__ __launch_bounds__(256) void gemm_bf16(const u16* __restrict__ A,
                                                 const u16* __restrict__ W,
                                                 u16* __restrict__ Cb,
                                                 float* __restrict__ Cf) {
    __shared__ u16 Bs[16 * BSTRIDE];
    const int tid     = threadIdx.x;
    const int colbase = blockIdx.x * 16;
    const int rowbase = blockIdx.y * 128;

    {   // async stage of the 16-channel weight panel (32 KB), ASYNCcnt-tracked
        const int c  = tid >> 4;          // channel 0..15
        const int k0 = (tid & 15) * 64;   // 64 contiguous u16 per thread
        const u16* wp = W + (size_t)(colbase + c) * HID + k0;
        const unsigned lbase = (unsigned)(size_t)(Bs + c * BSTRIDE + k0);
#pragma unroll
        for (int i = 0; i < 64; i += 8) {
            asm volatile("global_load_async_to_lds_b128 %0, %1, off"
                         :: "v"(lbase + i * 2),
                            "v"((unsigned long long)(size_t)(wp + i)) : "memory");
        }
        asm volatile("s_wait_asynccnt 0x0" ::: "memory");
    }
    __syncthreads();

    const int wave = tid >> 5;
    const int lane = tid & 31;
    const int c16  = lane & 15;   // M (A) / N (B) index within tile
    const int hl   = lane >> 4;   // K-half selector
    const u16* ap    = A + (size_t)(rowbase + wave * 16 + c16) * HID;
    const u16* bbase = Bs + c16 * BSTRIDE;

    v8f acc = {};
#pragma unroll 8
    for (int kk = 0; kk < HID; kk += 32) {
        Frag a, b;
        a.u4[0] = *(const uint4*)(ap + kk + hl * 8);
        a.u4[1] = *(const uint4*)(ap + kk + 16 + hl * 8);
        b.u4[0] = *(const uint4*)(bbase + kk + hl * 8);
        b.u4[1] = *(const uint4*)(bbase + kk + 16 + hl * 8);
        acc = __builtin_amdgcn_wmma_f32_16x16x32_bf16(false, a.v, false, b.v,
                                                      (short)0, acc, false, false);
    }

    const int col = colbase + c16;
#pragma unroll
    for (int r = 0; r < 8; ++r) {
        const int orow = rowbase + wave * 16 + r + 8 * hl;   // token index
        if (OUTK == OUTK_BF16) {
            Cb[(size_t)orow * HID + col] = f2bf(acc[r]);
        } else if (OUTK == OUTK_BF16_VT) {
            const int bb = orow >> 11, tt = orow & (SEQ - 1);
            const int hh = col >> 6,  dd = col & (HDIM - 1);
            Cb[(((size_t)bb * NHEAD + hh) * HDIM + dd) * SEQ + tt] = f2bf(acc[r]);
        } else {
            Cf[(size_t)orow * HID + col] = acc[r];
        }
    }
}

// ---------------------------------------------------------------------------
// 3) RoPE in-place on Q and K (bf16).  One thread owns a (d, d+32) pair.
//    Q additionally absorbs the softmax scale 1/sqrt(HDIM) = 0.125.
// ---------------------------------------------------------------------------
__global__ __launch_bounds__(256) void rope_kernel(u16* __restrict__ q,
                                                   u16* __restrict__ k) {
    const int idx = blockIdx.x * blockDim.x + threadIdx.x;   // NTOK*NHEAD*32
    const int j   = idx & 31;
    const int hh  = (idx >> 5) & (NHEAD - 1);
    const int row = idx >> 9;
    if (row >= NTOK) return;
    const int t = row & (SEQ - 1);
    const float inv = __expf(-(float)j * 0.2878231366f);     // ln(10000)/32
    float sn, cs;
    __sincosf((float)t * inv, &sn, &cs);
    const size_t base = (size_t)row * HID + hh * HDIM + j;
    {
        float x1 = bf2f(q[base]), x2 = bf2f(q[base + 32]);
        q[base]      = f2bf((x1 * cs - x2 * sn) * 0.125f);   // fold 1/sqrt(D)
        q[base + 32] = f2bf((x2 * cs + x1 * sn) * 0.125f);
    }
    {
        float x1 = bf2f(k[base]), x2 = bf2f(k[base + 32]);
        k[base]      = f2bf(x1 * cs - x2 * sn);
        k[base + 32] = f2bf(x2 * cs + x1 * sn);
    }
}

// ---------------------------------------------------------------------------
// 4) Flash attention.  Block = (b, h, 128 Q rows) = 8 waves x 16 rows.
//    K tiles: double-buffered in LDS via global_load_async_to_lds_b128;
//    V fragments: read directly from transposed-global Vt[b][h][d][t];
//    Q fragments: pinned in registers for all 32 KV tiles.
// ---------------------------------------------------------------------------
#define KVSTRIDE 72   // 64 + 8 u16 pad -> conflict-free ds_load_b128

__global__ __launch_bounds__(256) void attn_kernel(const u16* __restrict__ Q,
                                                   const u16* __restrict__ Kc,
                                                   const u16* __restrict__ Vt,
                                                   u16* __restrict__ O) {
    __shared__ u16 Ks[2][64 * KVSTRIDE];       // double-buffered K tile [key][d]
    __shared__ u16 Ps[8][16 * KVSTRIDE];       // per-wave P scratch

    const int tid  = threadIdx.x;
    const int qt   = blockIdx.x;               // 128-row Q tile
    const int h    = blockIdx.y;
    const int b    = blockIdx.z;
    const int wave = tid >> 5;
    const int lane = tid & 31;
    const int c16  = lane & 15;
    const int hl   = lane >> 4;

    // --- async K-tile staging: each thread moves two 16B chunks ------------
    const int lr  = tid >> 3;           // rows lr and lr+32 of the 64-key tile
    const int lc8 = (tid & 7) * 8;      // 8-u16 (16B) column chunk
    auto issue_tile = [&](int kt, int buf) {
        const u16* g0 = Kc + (size_t)(b * SEQ + kt + lr) * HID + h * HDIM + lc8;
        const u16* g1 = g0 + (size_t)32 * HID;
        unsigned l0 = (unsigned)(size_t)&Ks[buf][lr * KVSTRIDE + lc8];
        unsigned l1 = (unsigned)(size_t)&Ks[buf][(lr + 32) * KVSTRIDE + lc8];
        asm volatile("global_load_async_to_lds_b128 %0, %1, off"
                     :: "v"(l0), "v"((unsigned long long)(size_t)g0) : "memory");
        asm volatile("global_load_async_to_lds_b128 %0, %1, off"
                     :: "v"(l1), "v"((unsigned long long)(size_t)g1) : "memory");
    };

    // Q fragments: 16 rows x 64 (two K=32 steps), pinned in registers
    const int qrow = qt * 128 + wave * 16 + c16;
    const u16* qp  = Q + (size_t)(b * SEQ + qrow) * HID + h * HDIM;
    Frag qf[2];
#pragma unroll
    for (int ks = 0; ks < 2; ++ks) {
        qf[ks].u4[0] = *(const uint4*)(qp + ks * 32 + hl * 8);
        qf[ks].u4[1] = *(const uint4*)(qp + ks * 32 + 16 + hl * 8);
    }

    v8f o0 = {}, o1 = {}, o2 = {}, o3 = {};
    float m[8], l[8];
#pragma unroll
    for (int r = 0; r < 8; ++r) { m[r] = -1e30f; l[r] = 0.f; }

    const u16* vbase = Vt + ((size_t)(b * NHEAD + h) * HDIM) * SEQ;

    // prologue: fetch tile 0
    issue_tile(0, 0);
    asm volatile("s_wait_asynccnt 0x0" ::: "memory");
    __syncthreads();

    for (int kt = 0; kt < SEQ; kt += 64) {
        const int buf = (kt >> 6) & 1;
        if (kt + 64 < SEQ) issue_tile(kt + 64, buf ^ 1);   // overlap with compute

        // S = (Q/sqrt(D)) x K^T  (16 x 64); scale pre-folded into Q
        const u16* ksbase = Ks[buf];
        v8f s[4];
#pragma unroll
        for (int n = 0; n < 4; ++n) {
            v8f acc = {};
#pragma unroll
            for (int ks = 0; ks < 2; ++ks) {
                Frag kb;
                const u16* bp = ksbase + (n * 16 + c16) * KVSTRIDE + ks * 32 + hl * 8;
                kb.u4[0] = *(const uint4*)(bp);
                kb.u4[1] = *(const uint4*)(bp + 16);
                acc = __builtin_amdgcn_wmma_f32_16x16x32_bf16(false, qf[ks].v, false, kb.v,
                                                              (short)0, acc, false, false);
            }
            s[n] = acc;
        }

        // online softmax (row = element r + 8*hl; reduce across 16-lane half)
#pragma unroll
        for (int r = 0; r < 8; ++r) {
            float mx = fmaxf(fmaxf(s[0][r], s[1][r]), fmaxf(s[2][r], s[3][r]));
#pragma unroll
            for (int off = 8; off > 0; off >>= 1) mx = fmaxf(mx, __shfl_xor(mx, off, 32));
            const float mnew = fmaxf(m[r], mx);
            const float corr = __expf(m[r] - mnew);
            m[r] = mnew;
            l[r] *= corr;
            o0[r] *= corr; o1[r] *= corr; o2[r] *= corr; o3[r] *= corr;
            float rs = 0.f;
#pragma unroll
            for (int n = 0; n < 4; ++n) {
                const float p = __expf(s[n][r] - mnew);
                s[n][r] = p;
                rs += p;
            }
#pragma unroll
            for (int off = 8; off > 0; off >>= 1) rs += __shfl_xor(rs, off, 32);
            l[r] += rs;
        }

        // transpose P through per-wave LDS (D-layout -> A-layout)
        u16* pp = Ps[wave];
#pragma unroll
        for (int n = 0; n < 4; ++n)
#pragma unroll
            for (int r = 0; r < 8; ++r)
                pp[(r + 8 * hl) * KVSTRIDE + n * 16 + c16] = f2bf(s[n][r]);
        asm volatile("s_wait_dscnt 0x0" ::: "memory");

        Frag pf[2];
#pragma unroll
        for (int ks = 0; ks < 2; ++ks) {
            const u16* ap = pp + c16 * KVSTRIDE + ks * 32 + hl * 8;
            pf[ks].u4[0] = *(const uint4*)(ap);
            pf[ks].u4[1] = *(const uint4*)(ap + 16);
        }

        // O += P x V  (V B-fragments straight from global, contiguous in t)
#pragma unroll
        for (int d = 0; d < 4; ++d) {
            v8f* oc = (d == 0) ? &o0 : (d == 1) ? &o1 : (d == 2) ? &o2 : &o3;
#pragma unroll
            for (int ks = 0; ks < 2; ++ks) {
                Frag vb;
                const u16* vp = vbase + (size_t)(d * 16 + c16) * SEQ + kt + ks * 32 + hl * 8;
                vb.u4[0] = *(const uint4*)(vp);
                vb.u4[1] = *(const uint4*)(vp + 16);
                *oc = __builtin_amdgcn_wmma_f32_16x16x32_bf16(false, pf[ks].v, false, vb.v,
                                                              (short)0, *oc, false, false);
            }
        }

        asm volatile("s_wait_asynccnt 0x0" ::: "memory");   // tile kt+64 landed
        __syncthreads();
    }

    // normalize and store bf16 to O[NTOK][HID]
#pragma unroll
    for (int r = 0; r < 8; ++r) {
        const float inv = 1.f / l[r];
        const int orow  = b * SEQ + qt * 128 + wave * 16 + r + 8 * hl;
        u16* op = O + (size_t)orow * HID + h * HDIM + c16;
        op[0]  = f2bf(o0[r] * inv);
        op[16] = f2bf(o1[r] * inv);
        op[32] = f2bf(o2[r] * inv);
        op[48] = f2bf(o3[r] * inv);
    }
}

// ---------------------------------------------------------------------------
// Host side
// ---------------------------------------------------------------------------
extern "C" void kernel_launch(void* const* d_in, const int* in_sizes, int n_in,
                              void* d_out, int out_size, void* d_ws, size_t ws_size,
                              hipStream_t stream) {
    const float* x  = (const float*)d_in[0];
    const float* wq = (const float*)d_in[1];
    const float* wk = (const float*)d_in[2];
    const float* wv = (const float*)d_in[3];
    const float* wo = (const float*)d_in[4];

    char* ws = (char*)d_ws;
    size_t off = 0;
    u16* xb  = (u16*)(ws + off); off += (size_t)NTOK * HID * 2;  // 8 MB
    u16* wqb = (u16*)(ws + off); off += (size_t)HID * HID * 2;   // 2 MB
    u16* wkb = (u16*)(ws + off); off += (size_t)HID * HID * 2;
    u16* wvb = (u16*)(ws + off); off += (size_t)HID * HID * 2;
    u16* wob = (u16*)(ws + off); off += (size_t)HID * HID * 2;
    u16* qb  = (u16*)(ws + off); off += (size_t)NTOK * HID * 2;
    u16* kb  = (u16*)(ws + off); off += (size_t)NTOK * HID * 2;
    u16* vtb = (u16*)(ws + off); off += (size_t)NTOK * HID * 2;  // [B][H][D][T]
    u16* ob  = (u16*)(ws + off); off += (size_t)NTOK * HID * 2;  // total 48 MB

    cvt_f32_bf16<<<2048, 256, 0, stream>>>(x,  xb,  NTOK * HID);
    cvt_f32_bf16<<<1024, 256, 0, stream>>>(wq, wqb, HID * HID);
    cvt_f32_bf16<<<1024, 256, 0, stream>>>(wk, wkb, HID * HID);
    cvt_f32_bf16<<<1024, 256, 0, stream>>>(wv, wvb, HID * HID);
    cvt_f32_bf16<<<1024, 256, 0, stream>>>(wo, wob, HID * HID);

    dim3 ggrid(HID / 16, NTOK / 128);  // 64 x 32
    gemm_bf16<OUTK_BF16>   <<<ggrid, 256, 0, stream>>>(xb, wqb, qb,  nullptr);
    gemm_bf16<OUTK_BF16>   <<<ggrid, 256, 0, stream>>>(xb, wkb, kb,  nullptr);
    gemm_bf16<OUTK_BF16_VT><<<ggrid, 256, 0, stream>>>(xb, wvb, vtb, nullptr);

    rope_kernel<<<(NTOK * NHEAD * 32) / 256, 256, 0, stream>>>(qb, kb);

    attn_kernel<<<dim3(SEQ / 128, NHEAD, BATCH), 256, 0, stream>>>(qb, kb, vtb, ob);

    gemm_bf16<OUTK_F32><<<ggrid, 256, 0, stream>>>(ob, wob, nullptr, (float*)d_out);
}
